// GuuEncoder_64939905516200
// MI455X (gfx1250) — matrix operand order repaired and checked by reading, not compiled, not run on hardware
//
#include <hip/hip_runtime.h>
#include <hip/hip_bf16.h>
#include <stdint.h>

// ---------------- problem constants ----------------
#define VOCAB   50257
#define EMBED   512          // K of the GEMM
#define HID2    512          // N of the GEMM (HIDDEN/2)
#define NB      2048         // batch rows
#define SLEN    128          // tokens per row
#define MTOT    4096         // 2*NB GEMM rows (add-sums then rem-sums)
#define DDIM    1024         // combined / output dim

typedef __attribute__((ext_vector_type(16))) __bf16 v16bf;
typedef __attribute__((ext_vector_type(8)))  __bf16 v8bf;
typedef __attribute__((ext_vector_type(8)))  float  v8f;

// ---------------- small helpers ----------------
__device__ __forceinline__ uint16_t f32_to_bf16(float f) {
    uint32_t u = __builtin_bit_cast(uint32_t, f);
    uint32_t r = u + 0x7fffu + ((u >> 16) & 1u);   // round-to-nearest-even
    return (uint16_t)(r >> 16);
}

__device__ __forceinline__ uint32_t pcg(uint32_t v) {
    v = v * 747796405u + 2891336453u;
    uint32_t x = ((v >> ((v >> 28u) + 4u)) ^ v) * 277803737u;
    return (x >> 22u) ^ x;
}
__device__ __forceinline__ float u01(uint32_t h) {
    return (float)(h >> 8) * (1.0f / 16777216.0f);   // [0,1)
}
__device__ __forceinline__ float gaussf(uint32_t key) {
    float u1 = fmaxf(u01(pcg(key)), 1e-9f);
    float u2 = u01(pcg(key ^ 0x68BC21EBu));
    return sqrtf(-2.0f * __logf(u1)) * __cosf(6.28318530718f * u2);
}

// ---------------- kernel 0: w_prenoise f32 -> bf16 ----------------
__global__ void convert_b_kernel(const float* __restrict__ w, uint16_t* __restrict__ Bbf) {
    int i = (blockIdx.x * 256 + threadIdx.x) * 4;          // 512*512 / (256*4) = 256 blocks
    float4 v = *reinterpret_cast<const float4*>(w + i);
    uint32_t lo = (uint32_t)f32_to_bf16(v.x) | ((uint32_t)f32_to_bf16(v.y) << 16);
    uint32_t hi = (uint32_t)f32_to_bf16(v.z) | ((uint32_t)f32_to_bf16(v.w) << 16);
    uint2 p; p.x = lo; p.y = hi;
    *reinterpret_cast<uint2*>(Bbf + i) = p;
}

// ---------------- kernel 1: gather + sum over S, emit bf16 A (4096x512) ----------------
__global__ void gather_sum_kernel(const long long* __restrict__ added,
                                  const long long* __restrict__ removed,
                                  const float* __restrict__ embed,
                                  uint16_t* __restrict__ Abf) {
    const int rid = blockIdx.x;                // 0..4095
    const int set = rid >> 11;                 // 0 = added, 1 = removed
    const int b   = rid & (NB - 1);
    const long long* trow = (set ? removed : added) + (size_t)b * SLEN;
    const int tid = threadIdx.x;               // 256 threads, 2 floats each
    const int c0  = tid * 2;

    float acc0 = 0.0f, acc1 = 0.0f;
    int t = (int)trow[0];
    for (int s = 0; s < SLEN; ++s) {
        const float* erow = embed + (size_t)t * EMBED;
        float2 e = *reinterpret_cast<const float2*>(erow + c0);
        if (s + 1 < SLEN) {
            int tn = (int)trow[s + 1];
            __builtin_prefetch(embed + (size_t)tn * EMBED + c0, 0, 0);  // global_prefetch_b8
            t = tn;
        }
        acc0 += e.x; acc1 += e.y;
    }
    uint32_t packed = (uint32_t)f32_to_bf16(acc0) | ((uint32_t)f32_to_bf16(acc1) << 16);
    *reinterpret_cast<uint32_t*>(Abf + (size_t)rid * EMBED + c0) = packed;
}

// ---------------- kernel 2: bf16 WMMA GEMM (NT) with async-LDS double buffering -------
// A: (MTOT x 512) row-major bf16 ; B: (512 x 512) row-major bf16 (row = output col h)
// C[m][n] = sum_k A[m][k]*B[n][k]
// Block tile 64(M) x 128(N); per K-step stage A(64x32) + B(128x32) into LDS via
// global_load_async_to_lds_b128 (ASYNCcnt), double-buffered.
__global__ void gemm_wmma_kernel(const uint16_t* __restrict__ Abf,
                                 const uint16_t* __restrict__ Bbf,
                                 float* __restrict__ combined) {
    const int tid  = threadIdx.x;
    const int wave = tid >> 5;
    const int lane = tid & 31;
    const int wm = wave & 1;           // 2 wave-rows  (64 M per block)
    const int wn = wave >> 1;          // 4 wave-cols  (128 N per block)
    const int m_blk = blockIdx.x * 64;
    const int n_blk = blockIdx.y * 128;
    const int m0 = m_blk + wm * 32;
    const int n0 = n_blk + wn * 32;
    const int half = lane >> 4;
    const int mr   = lane & 15;

    // double-buffered LDS tiles
    __shared__ __align__(32) uint16_t sA[2][64 * 32];    // 2 x 4 KB
    __shared__ __align__(32) uint16_t sB[2][128 * 32];   // 2 x 8 KB

    // issue one K-step tile (per wave: 1 A-chunk + 2 B-chunks = 3 async ops)
    auto issue_tile = [&](int buf, int kb) {
        {   // A: 256 x 16B chunks, 1 per thread
            int c = tid, row = c >> 1, seg = c & 1;
            uint64_t g = (uint64_t)(uintptr_t)(Abf + (size_t)(m_blk + row) * EMBED + kb + seg * 8);
            uint32_t l = (uint32_t)(uintptr_t)(&sA[buf][row * 32 + seg * 8]);
            asm volatile("global_load_async_to_lds_b128 %0, %1, off" :: "v"(l), "v"(g) : "memory");
        }
#pragma unroll
        for (int i = 0; i < 2; ++i) {   // B: 512 x 16B chunks, 2 per thread
            int c = tid * 2 + i, row = c >> 2, seg = c & 3;
            uint64_t g = (uint64_t)(uintptr_t)(Bbf + (size_t)(n_blk + row) * EMBED + kb + seg * 8);
            uint32_t l = (uint32_t)(uintptr_t)(&sB[buf][row * 32 + seg * 8]);
            asm volatile("global_load_async_to_lds_b128 %0, %1, off" :: "v"(l), "v"(g) : "memory");
        }
    };

    v8f c00 = {}, c01 = {}, c10 = {}, c11 = {};

    issue_tile(0, 0);                       // prologue
    int cur = 0;
    for (int step = 0; step < EMBED / 32; ++step) {
        __syncthreads();                    // readers of buf cur^1 (step-1 compute) are done
        if (step + 1 < EMBED / 32) {
            issue_tile(cur ^ 1, (step + 1) * 32);
            // 3 ops of next tile may remain in flight; async loads complete in order,
            // so asynccnt<=3 guarantees this step's 3 ops have landed in LDS.
            asm volatile("s_wait_asynccnt 0x3" ::: "memory");
        } else {
            asm volatile("s_wait_asynccnt 0x0" ::: "memory");
        }
        __syncthreads();                    // all waves' current tile is visible

        const uint16_t* tA = &sA[cur][0];
        const uint16_t* tB = &sB[cur][0];

        // A fragments (ISA 16-bit A layout: lanes 0-15 K=0..7/16..23, lanes 16-31 K=8..15/24..31)
        v16bf a0, a1;
        {
            const int lrA0 = wm * 32 + mr;
            const int lrA1 = lrA0 + 16;
            v8bf lo0 = *reinterpret_cast<const v8bf*>(tA + lrA0 * 32 + half * 8);
            v8bf hi0 = *reinterpret_cast<const v8bf*>(tA + lrA0 * 32 + 16 + half * 8);
            v8bf lo1 = *reinterpret_cast<const v8bf*>(tA + lrA1 * 32 + half * 8);
            v8bf hi1 = *reinterpret_cast<const v8bf*>(tA + lrA1 * 32 + 16 + half * 8);
#pragma unroll
            for (int i = 0; i < 8; ++i) {
                a0[i] = lo0[i]; a0[i + 8] = hi0[i];
                a1[i] = lo1[i]; a1[i + 8] = hi1[i];
            }
        }
        // B fragments (same as A: lanes 0-15 K=0..7/16..23, lanes 16-31 K=8..15/24..31)
        const int lnB0 = wn * 32 + mr;
        const int lnB1 = lnB0 + 16;
        v16bf b0, b1;
        {
            v8bf blo0 = *reinterpret_cast<const v8bf*>(tB + lnB0 * 32 + half * 8);
            v8bf bhi0 = *reinterpret_cast<const v8bf*>(tB + lnB0 * 32 + 16 + half * 8);
            v8bf blo1 = *reinterpret_cast<const v8bf*>(tB + lnB1 * 32 + half * 8);
            v8bf bhi1 = *reinterpret_cast<const v8bf*>(tB + lnB1 * 32 + 16 + half * 8);
#pragma unroll
            for (int i = 0; i < 8; ++i) {
                b0[i] = blo0[i]; b0[i + 8] = bhi0[i];
                b1[i] = blo1[i]; b1[i + 8] = bhi1[i];
            }
        }

        c00 = __builtin_amdgcn_wmma_f32_16x16x32_bf16(false, a0, false, b0, (short)0, c00, false, false);
        c01 = __builtin_amdgcn_wmma_f32_16x16x32_bf16(false, a0, false, b1, (short)0, c01, false, false);
        c10 = __builtin_amdgcn_wmma_f32_16x16x32_bf16(false, a1, false, b0, (short)0, c10, false, false);
        c11 = __builtin_amdgcn_wmma_f32_16x16x32_bf16(false, a1, false, b1, (short)0, c11, false, false);

        cur ^= 1;
    }

    // C/D layout: VGPR r -> M = r (lanes 0-15) or M = r+8 (lanes 16-31); N = lane&15
    const int moff = half * 8;
    const int col0 = lane & 15;
#pragma unroll
    for (int t = 0; t < 4; ++t) {
        const v8f& c = (t == 0) ? c00 : (t == 1) ? c01 : (t == 2) ? c10 : c11;
        const int mt = m0 + ((t >> 1) * 16);
        const int nt = n0 + ((t & 1) * 16);
#pragma unroll
        for (int r = 0; r < 8; ++r) {
            int row = mt + moff + r;
            int col = nt + col0;
            float* dst = (row < NB)
                       ? (combined + (size_t)row * DDIM + col)
                       : (combined + (size_t)(row - NB) * DDIM + HID2 + col);
            *dst = c[r];
        }
    }
}

// ---------------- kernel 3: vMF sampling (one block per row) ----------------
__device__ __forceinline__ float sample_w_row(uint32_t row) {
    const float kap = 80.0f, dr = 1023.0f;
    const float bb = dr / (sqrtf(4.0f * kap * kap + dr * dr) + 2.0f * kap);
    const float xx = (1.0f - bb) / (1.0f + bb);
    const float cc = kap * xx + dr * __logf(1.0f - xx * xx);
    float w = 0.0f;
    bool done = false;
    for (uint32_t i = 0; i < 64u && !done; ++i) {
        uint32_t s = (row * 0x9E3779B9u) ^ (i * 0x85EBCA6Bu) ^ 0xC0FFEEu;
        // Beta(511.5,511.5) ~ N(0.5, 1/64)
        float z = 0.5f + gaussf(s) * 0.015625f;
        z = fminf(fmaxf(z, 1e-6f), 1.0f - 1e-6f);
        float wn = (1.0f - (1.0f + bb) * z) / (1.0f - (1.0f - bb) * z);
        float u  = fmaxf(u01(pcg(s ^ 0xDEADBEEFu)), 1e-12f);
        if (kap * wn + dr * __logf(fmaxf(1.0f - xx * wn, 1e-12f)) - cc >= __logf(u)) {
            w = wn; done = true;
        }
    }
    return w;
}

__global__ void vmf_sample_kernel(const float* __restrict__ combined,
                                  float* __restrict__ out) {
    const int row  = blockIdx.x;             // 0..2047
    const int tid  = threadIdx.x;            // 256 threads x 4 elems = 1024
    const int wave = tid >> 5;
    const int lane = tid & 31;
    const int base = tid * 4;

    float4 m4 = *reinterpret_cast<const float4*>(combined + (size_t)row * DDIM + base);
    float mu[4] = { m4.x, m4.y, m4.z, m4.w };
    float vr[4], rd[4];
#pragma unroll
    for (int j = 0; j < 4; ++j) {
        uint32_t idx = (uint32_t)(row * DDIM + base + j);
        vr[j] = gaussf(idx ^ 0x0BADF00Du);
        rd[j] = gaussf(idx ^ 0xFACEFEEDu);
    }

    // fused reduction: ||mu||^2, mu.vr, ||vr||^2, ||rd||^2
    float p0 = 0.f, p1 = 0.f, p2 = 0.f, p3 = 0.f;
#pragma unroll
    for (int j = 0; j < 4; ++j) {
        p0 += mu[j] * mu[j];
        p1 += mu[j] * vr[j];
        p2 += vr[j] * vr[j];
        p3 += rd[j] * rd[j];
    }
#pragma unroll
    for (int off = 16; off > 0; off >>= 1) {
        p0 += __shfl_xor(p0, off, 32);
        p1 += __shfl_xor(p1, off, 32);
        p2 += __shfl_xor(p2, off, 32);
        p3 += __shfl_xor(p3, off, 32);
    }
    __shared__ float red[8][4];
    if (lane == 0) { red[wave][0] = p0; red[wave][1] = p1; red[wave][2] = p2; red[wave][3] = p3; }
    __syncthreads();
    float s0 = 0.f, s1 = 0.f, s2 = 0.f, s3 = 0.f;
#pragma unroll
    for (int wv = 0; wv < 8; ++wv) { s0 += red[wv][0]; s1 += red[wv][1]; s2 += red[wv][2]; s3 += red[wv][3]; }

    const float munorm = sqrtf(s0);
    const float safe   = fmaxf(munorm, 1e-20f);
    const float proj   = s1 / safe;                              // mu_hat . v_raw
    const float orthon = sqrtf(fmaxf(s2 - proj * proj, 1e-20f)); // ||v_raw - mu_hat*proj||
    const float rdn    = sqrtf(fmaxf(s3, 1e-20f));

    const float trand   = u01(pcg((uint32_t)row ^ 0x7EA7A57Eu)) * 0.1f;     // NORM_EPS
    const float munoise = fminf(munorm, 13.9f) + trand;                     // clip + trand
    const float ufb     = u01(pcg((uint32_t)row ^ 0x5EEDFACEu)) * 0.1f;
    const float w       = sample_w_row((uint32_t)row);
    const float sw      = sqrtf(fmaxf(1.0f - w * w, 0.0f));
    const bool  use_main = (munorm > 1e-10f);

    float4 o;
    float res[4];
#pragma unroll
    for (int j = 0; j < 4; ++j) {
        float mh   = mu[j] / safe;
        float v    = (vr[j] - mh * proj) / orthon;
        float main = (v * sw + mh * w) * munoise;
        float fb   = (rd[j] / rdn) * ufb;
        res[j]     = use_main ? main : fb;
    }
    o.x = res[0]; o.y = res[1]; o.z = res[2]; o.w = res[3];
    *reinterpret_cast<float4*>(out + (size_t)row * DDIM + base) = o;
}

// ---------------- launcher ----------------
extern "C" void kernel_launch(void* const* d_in, const int* in_sizes, int n_in,
                              void* d_out, int out_size, void* d_ws, size_t ws_size,
                              hipStream_t stream) {
    const long long* added   = (const long long*)d_in[0];  // (2048,128) int64
    const long long* removed = (const long long*)d_in[1];  // (2048,128) int64
    const float*     embed   = (const float*)d_in[2];      // (50257,512) f32
    const float*     wpre    = (const float*)d_in[3];      // (512,512)   f32
    float*           out     = (float*)d_out;              // (2048,1024) f32

    uint8_t* ws = (uint8_t*)d_ws;
    uint16_t* Abf      = (uint16_t*)(ws);                  // 4096*512*2  = 4,194,304 B
    uint16_t* Bbf      = (uint16_t*)(ws + 4194304);        //  512*512*2  =   524,288 B
    float*    combined = (float*)   (ws + 4718592);        // 2048*1024*4 = 8,388,608 B

    convert_b_kernel <<<dim3(256),      dim3(256), 0, stream>>>(wpre, Bbf);
    gather_sum_kernel<<<dim3(MTOT),     dim3(256), 0, stream>>>(added, removed, embed, Abf);
    gemm_wmma_kernel <<<dim3(64, 4),    dim3(256), 0, stream>>>(Abf, Bbf, combined);
    vmf_sample_kernel<<<dim3(NB),       dim3(256), 0, stream>>>(combined, out);
}
